// GNNEmbeds_5987184411130
// MI455X (gfx1250) — matrix operand order repaired and checked by reading, not compiled, not run on hardware
//
#include <hip/hip_runtime.h>
#include <hip/hip_bf16.h>

typedef __attribute__((ext_vector_type(16))) _Float16 v16h;
typedef __attribute__((ext_vector_type(8)))  _Float16 v8h;
typedef __attribute__((ext_vector_type(8)))  float    v8f;

// ---------------------------------------------------------------------------
// Zero two float buffers (S0, S1 accumulators) in one launch.
// ---------------------------------------------------------------------------
__global__ void nnconv_zero2_kernel(float* __restrict__ a, float* __restrict__ b, int n) {
    int i = blockIdx.x * blockDim.x + threadIdx.x;
    if (i < n) { a[i] = 0.0f; b[i] = 0.0f; }
}

// ---------------------------------------------------------------------------
// Edge scatter: S0[dst] += x[src], S1[dst] += e * x[src]. One thread per
// (edge, feature) element; inC is a power of two (2 or 128) so the
// edge/feature split is shift+mask, no division.
// ---------------------------------------------------------------------------
__global__ void nnconv_scatter_kernel(const float* __restrict__ X,
                                      const int* __restrict__ src,
                                      const int* __restrict__ dst,
                                      const float* __restrict__ ea,
                                      float* __restrict__ S0,
                                      float* __restrict__ S1,
                                      int total, int fshift, int fmask) {
    int idx = blockIdx.x * blockDim.x + threadIdx.x;
    if (idx >= total) return;
    int e = idx >> fshift;
    int f = idx & fmask;
    int inC = fmask + 1;
    int s = src[e];
    int d = dst[e];
    float v = X[(size_t)s * inC + f];
    float w = ea[e];
    atomicAdd(&S0[(size_t)d * inC + f], v);
    atomicAdd(&S1[(size_t)d * inC + f], w * v);
}

// ---------------------------------------------------------------------------
// Pack activations: Hh[n, k] (f16, [N, Kpad]) = [ X | S0 | S1 | 0-pad ].
// 2-D grid: blockIdx.y = node row, x covers K -> no integer division.
// ---------------------------------------------------------------------------
__global__ void nnconv_packH_kernel(const float* __restrict__ X,
                                    const float* __restrict__ S0,
                                    const float* __restrict__ S1,
                                    _Float16* __restrict__ Hh,
                                    int inC, int Kpad) {
    int k = blockIdx.x * blockDim.x + threadIdx.x;
    int n = blockIdx.y;
    if (k >= Kpad) return;
    float v = 0.0f;
    if (k < inC)            v = X [(size_t)n * inC + k];
    else if (k < 2 * inC)   v = S0[(size_t)n * inC + (k - inC)];
    else if (k < 3 * inC)   v = S1[(size_t)n * inC + (k - 2 * inC)];
    Hh[(size_t)n * Kpad + k] = (_Float16)v;
}

// ---------------------------------------------------------------------------
// Pack weights transposed: WTh[o, k] (f16, [outC, Kpad]) row o =
// [ root[:,o] | A[:,o] | B[:,o] | 0-pad ] where A=lin_w.reshape(in,out),
// B=lin_b.reshape(in,out). Transposed so WMMA B-fragment loads are contiguous.
// ---------------------------------------------------------------------------
__global__ void nnconv_packW_kernel(const float* __restrict__ root,
                                    const float* __restrict__ lw,
                                    const float* __restrict__ lb,
                                    _Float16* __restrict__ WTh,
                                    int inC, int outC, int Kpad) {
    int k = blockIdx.x * blockDim.x + threadIdx.x;
    int o = blockIdx.y;
    if (k >= Kpad) return;
    float v = 0.0f;
    if (k < inC)            v = root[(size_t)k * outC + o];
    else if (k < 2 * inC)   v = lw[(size_t)(k - inC)     * outC + o];
    else if (k < 3 * inC)   v = lb[(size_t)(k - 2 * inC) * outC + o];
    WTh[(size_t)o * Kpad + k] = (_Float16)v;
}

// ---------------------------------------------------------------------------
// WMMA GEMM: Y[M, Nout] = Hh[M, Kpad] * WTh[Nout, Kpad]^T + bias (+ ReLU).
// Grid: blockIdx.y = M-tile, (blockIdx.x * 8 + waveId) = N-tile; 8 waves per
// 256-thread block. One wave32 per 16x16 output tile, K-loop in steps of 32
// using v_wmma_f32_16x16x32_f16 (f16 inputs, f32 accumulate).
// ---------------------------------------------------------------------------
__global__ void nnconv_wmma_gemm_kernel(const _Float16* __restrict__ Hh,
                                        const _Float16* __restrict__ WTh,
                                        const float* __restrict__ bias,
                                        float* __restrict__ Y,
                                        int Kpad, int Nout, int relu) {
    int wave = threadIdx.x >> 5;                 // 0..7
    int lane = threadIdx.x & 31;
    int tn = blockIdx.x * 8 + wave;              // tile along N (out features)
    int tm = blockIdx.y;                         // tile along M (nodes)
    int tilesN = Nout >> 4;
    if (tn >= tilesN) return;                    // wave-uniform: EXEC all-1s at WMMA

    int rsel  = lane & 15;
    int khalf = lane >> 4;                       // 0: lanes 0-15, 1: lanes 16-31

    // A-fragment (16x32 f16): lanes 0-15 row M=rsel holds K k0+0..7 (v0-3) and
    // k0+16..23 (v4-7); lanes 16-31 hold K k0+8..15 and k0+24..31.
    const _Float16* ar = Hh  + (size_t)(tm * 16 + rsel) * Kpad + khalf * 8;
    // B-fragment (32x16 f16): col N=rsel, same K split as A: lanes 0-15 hold
    // K k0+0..7 and k0+16..23; lanes 16-31 hold K k0+8..15 and k0+24..31.
    const _Float16* br = WTh + (size_t)(tn * 16 + rsel) * Kpad + khalf * 8;

    v8f acc = {};
    for (int k0 = 0; k0 < Kpad; k0 += 32) {
        union { v16h v; v8h h[2]; } a;
        a.h[0] = *(const v8h*)(ar + k0);
        a.h[1] = *(const v8h*)(ar + k0 + 16);
        union { v16h v; v8h h[2]; } bu;
        bu.h[0] = *(const v8h*)(br + k0);
        bu.h[1] = *(const v8h*)(br + k0 + 16);
        v16h b = bu.v;
        acc = __builtin_amdgcn_wmma_f32_16x16x32_f16(
            /*neg_a=*/false, a.v, /*neg_b=*/false, b,
            /*c_mod=*/(short)0, acc, /*reuse_a=*/false, /*reuse_b=*/false);
    }

    // C/D layout: VGPR r -> row r + 8*khalf, col = rsel (within tile).
    int row0 = tm * 16 + khalf * 8;
    int col  = tn * 16 + rsel;
    float bs = bias[col];
#pragma unroll
    for (int r = 0; r < 8; ++r) {
        float v = acc[r] + bs;
        if (relu) v = v > 0.0f ? v : 0.0f;
        Y[(size_t)(row0 + r) * Nout + col] = v;
    }
}

// ---------------------------------------------------------------------------
// Host side
// ---------------------------------------------------------------------------
extern "C" void kernel_launch(void* const* d_in, const int* in_sizes, int n_in,
                              void* d_out, int out_size, void* d_ws, size_t ws_size,
                              hipStream_t stream) {
    const float* x    = (const float*)d_in[0];
    const int*   eidx = (const int*)  d_in[1];
    const float* ea   = (const float*)d_in[2];
    const float* lw[3]   = { (const float*)d_in[3],  (const float*)d_in[7],  (const float*)d_in[11] };
    const float* lb[3]   = { (const float*)d_in[4],  (const float*)d_in[8],  (const float*)d_in[12] };
    const float* root[3] = { (const float*)d_in[5],  (const float*)d_in[9],  (const float*)d_in[13] };
    const float* bias[3] = { (const float*)d_in[6],  (const float*)d_in[10], (const float*)d_in[14] };

    const int E   = in_sizes[1] / 2;
    const int H   = in_sizes[6];            // 128
    const int IN  = in_sizes[3] / H;        // 2
    const int N   = in_sizes[0] / IN;       // 1024
    const int OUT = in_sizes[14];           // 128

    const int* src = eidx;
    const int* dst = eidx + E;

    // Workspace layout (256B aligned). Max Kpad = 3*H = 384. Total ~2.9 MB.
    const int KpadMax = ((3 * H + 31) / 32) * 32;
    char* ws = (char*)d_ws;
    size_t off = 0;
    auto alloc = [&](size_t bytes) -> char* {
        char* p = ws + off;
        off = (off + bytes + 255) & ~(size_t)255;
        return p;
    };
    float*     S0  = (float*)    alloc((size_t)N * H * sizeof(float));
    float*     S1  = (float*)    alloc((size_t)N * H * sizeof(float));
    _Float16*  Hh  = (_Float16*) alloc((size_t)N * KpadMax * sizeof(_Float16));
    _Float16*  WTh = (_Float16*) alloc((size_t)H * KpadMax * sizeof(_Float16)); // outC<=128 rows
    float*     X2  = (float*)    alloc((size_t)N * H * sizeof(float));
    float*     X3  = (float*)    alloc((size_t)N * H * sizeof(float));
    (void)ws_size;

    const int BLK = 256;
    auto cdiv = [](int a, int b) { return (a + b - 1) / b; };

    auto run_layer = [&](const float* Xin, int inC, int li, int outC,
                         float* Yout, int relu) {
        const int Kpad = ((3 * inC + 31) / 32) * 32;
        const int nS = N * inC;
        nnconv_zero2_kernel<<<cdiv(nS, BLK), BLK, 0, stream>>>(S0, S1, nS);

        const int fshift = __builtin_ctz(inC);   // inC is a power of two
        const int nScat = E * inC;
        nnconv_scatter_kernel<<<cdiv(nScat, BLK), BLK, 0, stream>>>(
            Xin, src, dst, ea, S0, S1, nScat, fshift, inC - 1);

        dim3 gH(cdiv(Kpad, BLK), N);
        nnconv_packH_kernel<<<gH, BLK, 0, stream>>>(Xin, S0, S1, Hh, inC, Kpad);

        dim3 gW(cdiv(Kpad, BLK), outC);
        nnconv_packW_kernel<<<gW, BLK, 0, stream>>>(
            root[li], lw[li], lb[li], WTh, inC, outC, Kpad);

        dim3 gG(cdiv(outC / 16, 8), N / 16);     // 8 waves (N-tiles) per block
        nnconv_wmma_gemm_kernel<<<gG, BLK, 0, stream>>>(
            Hh, WTh, bias[li], Yout, Kpad, outC, relu);
    };

    run_layer(x,  IN, 0, H,   X2,            /*relu=*/1);
    run_layer(X2, H,  1, H,   X3,            /*relu=*/1);
    run_layer(X3, H,  2, OUT, (float*)d_out, /*relu=*/0);
}
